// MosaicBlockLayer_64707977281542
// MI455X (gfx1250) — matrix operand order repaired and checked by reading, not compiled, not run on hardware
//
#include <hip/hip_runtime.h>
#include <hip/hip_bf16.h>

// MosaicBlockLayer for MI455X (gfx1250, wave32).
// Big GEMMs: v_wmma_f32_16x16x32_bf16 (fp32 accumulate) fed from LDS tiles
// filled by CDNA5 async global->LDS copies (ASYNCcnt), double buffered.
// Sequential scans (SSM, hash memory) are latency-bound scalar kernels.
// All launches on `stream`, no sync calls (graph-capture safe).

typedef __attribute__((ext_vector_type(16))) __bf16 v16bf;
typedef __attribute__((ext_vector_type(8)))  __bf16 v8bf;
typedef __attribute__((ext_vector_type(8)))  float  v8f;
typedef __attribute__((ext_vector_type(4)))  int    v4i;

#define DEV static __device__ __forceinline__

DEV float sigmoidf_(float x) { return 1.0f / (1.0f + __expf(-x)); }
DEV float siluf_(float x)    { return x * sigmoidf_(x); }

DEV float wred32(float v) {
#pragma unroll
  for (int o = 16; o; o >>= 1) v += __shfl_xor(v, o, 32);
  return v;
}

// ---------------------------------------------------------------------------
// CDNA5 async global->LDS copy path (ASYNCcnt). Builtin signature (probed via
// the round-2 diagnostic): (v4i AS1*, v4i AS3*, imm offset, imm cpol).
// Guarded so the file still compiles if a toolchain lacks the builtins
// (falls back to sync copies with the identical double-buffer control flow).
// ---------------------------------------------------------------------------
#if __has_builtin(__builtin_amdgcn_global_load_async_to_lds_b128)
#define GEMM_ASYNC 1
#define CP16(gp, lp)                                                     \
  __builtin_amdgcn_global_load_async_to_lds_b128(                        \
      (__attribute__((address_space(1))) v4i*)(gp),                      \
      (__attribute__((address_space(3))) v4i*)(lp), 0, 0)
#if __has_builtin(__builtin_amdgcn_s_wait_asynccnt)
#define AWAIT6() __builtin_amdgcn_s_wait_asynccnt(6)
#define AWAIT0() __builtin_amdgcn_s_wait_asynccnt(0)
#else
#define AWAIT6() asm volatile("s_wait_asynccnt 0x6" ::: "memory")
#define AWAIT0() asm volatile("s_wait_asynccnt 0x0" ::: "memory")
#endif
#else
#define GEMM_ASYNC 0
#define CP16(gp, lp) do { *(v8bf*)(lp) = *(const v8bf*)(gp); } while (0)
#define AWAIT6() do {} while (0)
#define AWAIT0() do {} while (0)
#endif

// Fragment loads built from two 16B halves (avoids 32B alignment demands).
// A frag (16x32, ISA 7.12.2): lane = M row (lane&15), K chunks
// {klo..klo+7, klo+16..klo+23}, klo=(lane>>4)*8.
DEV v16bf frag_a(const __bf16* p) {
  union { v16bf v; v8bf h[2]; } u;
  u.h[0] = *(const v8bf*)(p);
  u.h[1] = *(const v8bf*)(p + 16);
  return u.v;
}
// B frag (32x16): lane = N col (lane&15), K chunks {klo..klo+7, klo+16..klo+23}, klo=(lane>>4)*8.
DEV v16bf frag_b(const __bf16* p) {
  union { v16bf v; v8bf h[2]; } u;
  u.h[0] = *(const v8bf*)(p);
  u.h[1] = *(const v8bf*)(p + 16);
  return u.v;
}

// ---------------------------------------------------------------------------
// fp32 -> bf16 convert (weights + activations staging)
// ---------------------------------------------------------------------------
__global__ __launch_bounds__(256) void k_f2bf(const float* __restrict__ in,
                                              __bf16* __restrict__ out, int n) {
  int i = blockIdx.x * 256 + threadIdx.x;
  if (i < n) out[i] = (__bf16)in[i];
}

// ---------------------------------------------------------------------------
// RMSNorm over rows of 1024, optional fp32 and bf16 outputs
// ---------------------------------------------------------------------------
__global__ __launch_bounds__(256) void k_rmsnorm(const float* __restrict__ x,
                                                 float* __restrict__ outf,
                                                 __bf16* __restrict__ outb) {
  __shared__ float red[8];
  const int m = blockIdx.x, tid = threadIdx.x;
  const float* row = x + (size_t)m * 1024;
  float ss = 0.f;
#pragma unroll 4
  for (int i = tid; i < 1024; i += 256) { float v = row[i]; ss = fmaf(v, v, ss); }
  ss = wred32(ss);
  if ((tid & 31) == 0) red[tid >> 5] = ss;
  __syncthreads();
  float tot = 0.f;
#pragma unroll
  for (int i = 0; i < 8; i++) tot += red[i];
  const float inv = rsqrtf(tot * (1.0f / 1024.0f) + 1e-6f);
  for (int i = tid; i < 1024; i += 256) {
    float v = row[i] * inv;
    size_t o = (size_t)m * 1024 + i;
    if (outf) outf[o] = v;
    if (outb) outb[o] = (__bf16)v;
  }
}

// ---------------------------------------------------------------------------
// WMMA GEMM:  C[m,n] = act( sum_k A[m,k] * W[n,k] + bias[n] ) + addend[m,n]
// A: M x K bf16 row-major, W: N x K bf16 row-major (C = A @ W^T).
// 8 waves / block, 64(M) x 128(N) macro tile, wave tile 32x32, K-tile 64.
// LDS double buffer filled with async global->LDS b128 copies; row stride
// padded to 72 bf16 (144B) for conflict-light ds_load_b128 fragment reads.
// act: 0 = identity, 1 = silu.
// ---------------------------------------------------------------------------
#define LDT 72                      // padded LDS tile row stride (bf16 elems)
#define TILE_ELEMS ((64 + 128) * LDT)

__global__ __launch_bounds__(256) void k_wmma_gemm(
    const __bf16* __restrict__ A, const __bf16* __restrict__ W,
    const float* __restrict__ bias, const float* __restrict__ addend,
    float* __restrict__ C, __bf16* __restrict__ Cbf,
    int M, int N, int K, int act) {
  __shared__ __bf16 smem[2][TILE_ELEMS];  // [A 64xLDT][B 128xLDT] per buffer
  const int tid  = threadIdx.x;
  const int lane = tid & 31;
  const int wave = tid >> 5;
  const int wm = wave & 1;               // 2 waves over M
  const int wn = wave >> 1;              // 4 waves over N
  const int mblk = blockIdx.y * 64;
  const int nblk = blockIdx.x * 128;

  // Per-thread cooperative fill: A tile 64x64 = 512 16B-chunks (2/thread),
  // B tile 128x64 = 1024 16B-chunks (4/thread). 6 async issues per tile.
  auto issue_tile = [&](int kt, int buf) {
    const int k0 = kt * 64;
    __bf16* As = &smem[buf][0];
    __bf16* Bs = &smem[buf][64 * LDT];
#pragma unroll
    for (int i = 0; i < 2; i++) {
      const int c = tid + i * 256;
      const int r = c >> 3, ch = (c & 7) * 8;
      CP16(A + (size_t)(mblk + r) * K + k0 + ch, As + r * LDT + ch);
    }
#pragma unroll
    for (int i = 0; i < 4; i++) {
      const int c = tid + i * 256;
      const int r = c >> 3, ch = (c & 7) * 8;
      CP16(W + (size_t)(nblk + r) * K + k0 + ch, Bs + r * LDT + ch);
    }
  };

  const int ntiles = K >> 6;
  issue_tile(0, 0);

  v8f acc[2][2] = {};
  const int arow = wm * 32 + (lane & 15);
  const int brow = wn * 32 + (lane & 15);
  const int aklo = (lane >> 4) * 8;
  const int bklo = (lane >> 4) * 8;

  for (int kt = 0; kt < ntiles; ++kt) {
    const int buf = kt & 1;
    if (kt + 1 < ntiles) { issue_tile(kt + 1, buf ^ 1); AWAIT6(); }
    else                 { AWAIT0(); }
    __syncthreads();     // tile kt visible from every wave
    const __bf16* As = &smem[buf][0];
    const __bf16* Bs = &smem[buf][64 * LDT];
#pragma unroll
    for (int kk = 0; kk < 64; kk += 32) {
      v16bf fa0 = frag_a(As + (arow)      * LDT + kk + aklo);
      v16bf fa1 = frag_a(As + (arow + 16) * LDT + kk + aklo);
      v16bf fb0 = frag_b(Bs + (brow)      * LDT + kk + bklo);
      v16bf fb1 = frag_b(Bs + (brow + 16) * LDT + kk + bklo);
      acc[0][0] = __builtin_amdgcn_wmma_f32_16x16x32_bf16(
          false, fa0, false, fb0, (short)0, acc[0][0], false, false);
      acc[0][1] = __builtin_amdgcn_wmma_f32_16x16x32_bf16(
          false, fa0, false, fb1, (short)0, acc[0][1], false, false);
      acc[1][0] = __builtin_amdgcn_wmma_f32_16x16x32_bf16(
          false, fa1, false, fb0, (short)0, acc[1][0], false, false);
      acc[1][1] = __builtin_amdgcn_wmma_f32_16x16x32_bf16(
          false, fa1, false, fb1, (short)0, acc[1][1], false, false);
    }
    __syncthreads();     // reads done before next tile overwrites buf
  }

  // C/D layout: lane 0-15 -> N=lane, M = vgpr r; lane 16-31 -> M = r+8
  const int m0 = mblk + wm * 32;
  const int n0 = nblk + wn * 32;
  const int colb  = lane & 15;
  const int rhalf = (lane >> 4) * 8;
#pragma unroll
  for (int i = 0; i < 2; i++) {
#pragma unroll
    for (int j = 0; j < 2; j++) {
      const int col = n0 + j * 16 + colb;
      const float bv = bias ? bias[col] : 0.f;
#pragma unroll
      for (int r = 0; r < 8; r++) {
        const int mr = m0 + i * 16 + rhalf + r;
        float v = acc[i][j][r] + bv;
        if (act == 1) v = siluf_(v);
        const size_t o = (size_t)mr * N + col;
        if (addend) v += addend[o];
        if (C)   C[o]   = v;
        if (Cbf) Cbf[o] = (__bf16)v;
      }
    }
  }
}

// ---------------------------------------------------------------------------
// u = silu(causal_conv4(xn)) * sigmoid(gate_lin + gate_b)
// ---------------------------------------------------------------------------
__global__ __launch_bounds__(256) void k_u(const float* __restrict__ xn,
                                           const float* __restrict__ gate_lin,
                                           const float* __restrict__ gate_b,
                                           const float* __restrict__ conv_w,
                                           float* __restrict__ uf,
                                           __bf16* __restrict__ ub) {
  const size_t idx = (size_t)blockIdx.x * 256 + threadIdx.x;  // 2048*1024
  const int dch = (int)(idx & 1023);
  const int m   = (int)(idx >> 10);
  const int b   = m >> 9;
  const int t   = m & 511;
  float h = 0.f;
#pragma unroll
  for (int j = 0; j < 4; j++) {
    const int tt = t - 3 + j;
    if (tt >= 0)
      h = fmaf(xn[((size_t)(b * 512 + tt) << 10) + dch], conv_w[dch * 4 + j], h);
  }
  const float g = sigmoidf_(gate_lin[idx] + gate_b[dch]);
  const float uu = siluf_(h) * g;
  uf[idx] = uu;
  ub[idx] = (__bf16)uu;
}

// ---------------------------------------------------------------------------
// SSM decay scan: s = sigmoid(logit[k]) * s + inp, over T. 16384 channels.
// ---------------------------------------------------------------------------
__global__ __launch_bounds__(256) void k_ssm(const float* __restrict__ inp,
                                             const float* __restrict__ decay_logit,
                                             __bf16* __restrict__ s_out) {
  const int c   = blockIdx.x * 256 + threadIdx.x;  // 0..16383
  const int b   = c >> 12;
  const int rem = c & 4095;
  const int k   = rem >> 10;
  const float decay = sigmoidf_(decay_logit[k]);
  float s = 0.f;
  size_t o = (size_t)b * 512 * 4096 + rem;
  for (int t = 0; t < 512; ++t) {
    s = fmaf(decay, s, inp[o]);
    s_out[o] = (__bf16)s;
    o += 4096;
  }
}

// ---------------------------------------------------------------------------
// Skinny projections from u: 135 output dims total. One block per token row,
// wave-parallel dot products with shfl reduction (WMMA is wasted at N=135).
// ---------------------------------------------------------------------------
__global__ __launch_bounds__(256) void k_heads(
    const float* __restrict__ u,
    const float* __restrict__ rbw, const float* __restrict__ wbw,
    const float* __restrict__ qkw, const float* __restrict__ wkw,
    const float* __restrict__ vw,
    const float* __restrict__ wgw, const float* __restrict__ wgb,
    const float* __restrict__ glw, const float* __restrict__ glb,
    const float* __restrict__ gmw, const float* __restrict__ gmb,
    float* __restrict__ rb, float* __restrict__ wb,
    float* __restrict__ qk, float* __restrict__ wk, float* __restrict__ wvo,
    float* __restrict__ wg, float* __restrict__ gl, float* __restrict__ gm) {
  __shared__ float ur[1024];
  const int m = blockIdx.x;
  for (int i = threadIdx.x; i < 1024; i += 256) ur[i] = u[(size_t)m * 1024 + i];
  __syncthreads();
  const int wave = threadIdx.x >> 5, lane = threadIdx.x & 31;
  for (int j = wave; j < 135; j += 8) {
    const float* wrow;
    float bias = 0.f;
    int kind, idx = 0;
    if (j < 18)       { wrow = rbw + (size_t)j * 1024;          kind = 0; idx = j; }
    else if (j < 36)  { wrow = wbw + (size_t)(j - 18) * 1024;   kind = 1; idx = j - 18; }
    else if (j < 68)  { wrow = qkw + (size_t)(j - 36) * 1024;   kind = 2; idx = j - 36; }
    else if (j < 100) { wrow = wkw + (size_t)(j - 68) * 1024;   kind = 3; idx = j - 68; }
    else if (j < 132) { wrow = vw  + (size_t)(j - 100) * 1024;  kind = 4; idx = j - 100; }
    else if (j == 132){ wrow = wgw; bias = wgb[0];              kind = 5; }
    else if (j == 133){ wrow = glw; bias = glb[0];              kind = 6; }
    else              { wrow = gmw; bias = gmb[0];              kind = 7; }
    float p = 0.f;
    for (int k = lane; k < 1024; k += 32) p = fmaf(ur[k], wrow[k], p);
    p = wred32(p);
    if (lane == 0) {
      p += bias;
      switch (kind) {
        case 0: rb[(size_t)m * 18 + idx] = p; break;
        case 1: wb[(size_t)m * 18 + idx] = p; break;
        case 2: qk[(size_t)m * 32 + idx] = p; break;
        case 3: wk[(size_t)m * 32 + idx] = p; break;
        case 4: wvo[(size_t)m * 32 + idx] = p; break;
        case 5: wg[m] = sigmoidf_(p); break;
        case 6: gl[m] = sigmoidf_(p); break;
        case 7: gm[m] = sigmoidf_(p); break;
      }
    }
  }
}

// ---------------------------------------------------------------------------
// Hash-memory state init (workspace not re-poisoned between replays)
// ---------------------------------------------------------------------------
__global__ __launch_bounds__(256) void k_meminit(float* __restrict__ memkv,
                                                 int* __restrict__ memlast) {
  const int i = blockIdx.x * 256 + threadIdx.x;
  if (i < 262144) memkv[i] = 0.f;
  if (i < 4096)   memlast[i] = -1;
}

// ---------------------------------------------------------------------------
// Hash-memory sequential scan. 1 block, 8 waves; wave = (b,h), lane = dim.
// B=4 H=2 BUCKETS=512 ASSOC=2 KEY=MEM=32, softmax over H*ASSOC=4 per batch.
// ---------------------------------------------------------------------------
__global__ __launch_bounds__(256) void k_memscan(
    const float* __restrict__ rb, const float* __restrict__ wb,
    const float* __restrict__ qk, const float* __restrict__ wk,
    const float* __restrict__ wv, const float* __restrict__ wg,
    float* __restrict__ mem_k, float* __restrict__ mem_v,
    int* __restrict__ mem_last, float* __restrict__ reads) {
  const int tid = threadIdx.x;
  const int wave = tid >> 5, lane = tid & 31;
  const int b = wave >> 1, h = wave & 1;
  __shared__ float sc[4][2][2];
  __shared__ float rp[4][2][32];
  const float scale = 0.1767766952966369f;  // 1/sqrt(32)
  for (int t = 0; t < 512; ++t) {
    const int m = b * 512 + t;
    int ri = 0, wi = 0;
#pragma unroll
    for (int j = 0; j < 9; ++j) {
      if (rb[(size_t)m * 18 + h * 9 + j] > 0.f) ri |= (1 << j);
      if (wb[(size_t)m * 18 + h * 9 + j] > 0.f) wi |= (1 << j);
    }
    const size_t bh = (size_t)(b * 2 + h);
    const float q = qk[(size_t)m * 32 + lane];
    const float* kp = mem_k + ((bh * 512 + ri) * 2) * 32;
    float p0 = wred32(kp[lane] * q);
    float p1 = wred32(kp[32 + lane] * q);
    if (lane == 0) { sc[b][h][0] = p0 * scale; sc[b][h][1] = p1 * scale; }
    __syncthreads();
    const float s00 = sc[b][0][0], s01 = sc[b][0][1];
    const float s10 = sc[b][1][0], s11 = sc[b][1][1];
    const float mx = fmaxf(fmaxf(s00, s01), fmaxf(s10, s11));
    const float e00 = __expf(s00 - mx), e01 = __expf(s01 - mx);
    const float e10 = __expf(s10 - mx), e11 = __expf(s11 - mx);
    const float inv = 1.f / (e00 + e01 + e10 + e11);
    const float a0 = (h == 0 ? e00 : e10) * inv;
    const float a1 = (h == 0 ? e01 : e11) * inv;
    const float* vp = mem_v + ((bh * 512 + ri) * 2) * 32;
    rp[b][h][lane] = a0 * vp[lane] + a1 * vp[32 + lane];
    __syncthreads();
    if (h == 0)
      reads[(size_t)m * 32 + lane] = rp[b][0][lane] + rp[b][1][lane];
    // write phase (own (b,h) region only; reads of mem_v already done)
    const size_t li = bh * 512 + wi;
    const int slot = (mem_last[li] + 1) & 1;
    const float g = wg[m];
    float* kw  = mem_k + (li * 2 + slot) * 32;
    float* vw2 = mem_v + (li * 2 + slot) * 32;
    const float ok = kw[lane], ov = vw2[lane];
    kw[lane]  = fmaf(g, wk[(size_t)m * 32 + lane] - ok, ok);
    vw2[lane] = fmaf(g, wv[(size_t)m * 32 + lane] - ov, ov);
    if (lane == 0) mem_last[li] = slot;
    __syncthreads();
  }
}

// ---------------------------------------------------------------------------
// y = x + gl*long + gm*(reads @ mem_out_w^T)
// ---------------------------------------------------------------------------
__global__ __launch_bounds__(256) void k_ycomb(
    const float* __restrict__ x, const float* __restrict__ longv,
    const float* __restrict__ reads, const float* __restrict__ mem_out_w,
    const float* __restrict__ gl, const float* __restrict__ gm,
    float* __restrict__ y) {
  const size_t idx = (size_t)blockIdx.x * 256 + threadIdx.x;
  const int dch = (int)(idx & 1023);
  const size_t m = idx >> 10;
  const float* rr = reads + m * 32;
  const float* wr = mem_out_w + (size_t)dch * 32;
  float mr = 0.f;
#pragma unroll
  for (int r = 0; r < 32; r++) mr = fmaf(rr[r], wr[r], mr);
  y[idx] = x[idx] + gl[m] * longv[idx] + gm[m] * mr;
}

// ---------------------------------------------------------------------------
extern "C" void kernel_launch(void* const* d_in, const int* in_sizes, int n_in,
                              void* d_out, int out_size, void* d_ws, size_t ws_size,
                              hipStream_t stream) {
  const float* x           = (const float*)d_in[0];
  const float* conv_w      = (const float*)d_in[1];
  const float* gate_w      = (const float*)d_in[2];
  const float* gate_b      = (const float*)d_in[3];
  const float* mlp_up_w    = (const float*)d_in[4];
  const float* mlp_up_b    = (const float*)d_in[5];
  const float* mlp_down_w  = (const float*)d_in[6];
  const float* mlp_down_b  = (const float*)d_in[7];
  const float* state_in_w  = (const float*)d_in[8];
  const float* state_out_w = (const float*)d_in[9];
  const float* decay_logit = (const float*)d_in[10];
  const float* read_bits_w = (const float*)d_in[11];
  const float* write_bits_w= (const float*)d_in[12];
  const float* qkey_w      = (const float*)d_in[13];
  const float* wkey_w      = (const float*)d_in[14];
  const float* value_w     = (const float*)d_in[15];
  const float* mem_out_w   = (const float*)d_in[16];
  const float* write_gate_w= (const float*)d_in[17];
  const float* write_gate_b= (const float*)d_in[18];
  const float* gate_long_w = (const float*)d_in[19];
  const float* gate_long_b = (const float*)d_in[20];
  const float* gate_mem_w  = (const float*)d_in[21];
  const float* gate_mem_b  = (const float*)d_in[22];

  const int M = 2048, D = 1024, KD = 4096, HID = 4096;

  char* base = (char*)d_ws;
  size_t off = 0;
  auto alloc = [&](size_t bytes) -> void* {
    void* p = base + off;
    off = (off + bytes + 255) & ~(size_t)255;
    return p;
  };
  float*  xn_f   = (float*) alloc((size_t)M * D * 4);
  __bf16* xn_b   = (__bf16*)alloc((size_t)M * D * 2);
  float*  gate_l = (float*) alloc((size_t)M * D * 4);
  float*  u_f    = (float*) alloc((size_t)M * D * 4);
  __bf16* u_b    = (__bf16*)alloc((size_t)M * D * 2);
  float*  inp_f  = (float*) alloc((size_t)M * KD * 4);
  __bf16* s_b    = (__bf16*)alloc((size_t)M * KD * 2);
  float*  long_f = (float*) alloc((size_t)M * D * 4);
  float*  y_f    = (float*) alloc((size_t)M * D * 4);
  __bf16* yn_b   = (__bf16*)alloc((size_t)M * D * 2);
  __bf16* t1_b   = (__bf16*)alloc((size_t)M * HID * 2);
  __bf16* gw_b   = (__bf16*)alloc((size_t)D * D * 2);
  __bf16* siw_b  = (__bf16*)alloc((size_t)KD * D * 2);
  __bf16* sow_b  = (__bf16*)alloc((size_t)D * KD * 2);
  __bf16* upw_b  = (__bf16*)alloc((size_t)HID * D * 2);
  __bf16* dnw_b  = (__bf16*)alloc((size_t)D * HID * 2);
  float*  rb_f   = (float*) alloc((size_t)M * 18 * 4);
  float*  wb_f   = (float*) alloc((size_t)M * 18 * 4);
  float*  qk_f   = (float*) alloc((size_t)M * 32 * 4);
  float*  wk_f   = (float*) alloc((size_t)M * 32 * 4);
  float*  wv_f   = (float*) alloc((size_t)M * 32 * 4);
  float*  wg_f   = (float*) alloc((size_t)M * 4);
  float*  gl_f   = (float*) alloc((size_t)M * 4);
  float*  gm_f   = (float*) alloc((size_t)M * 4);
  float*  reads  = (float*) alloc((size_t)M * 32 * 4);
  float*  memkv  = (float*) alloc((size_t)262144 * 4);   // mem_k ++ mem_v
  int*    memlast= (int*)   alloc((size_t)4096 * 4);
  if (off > ws_size) return;  // workspace too small: do nothing (deterministic)
  float* mem_k = memkv;
  float* mem_v = memkv + 131072;

  const dim3 blk(256);
  auto cvt = [&](const float* src, __bf16* dst, int n) {
    k_f2bf<<<dim3((n + 255) / 256), blk, 0, stream>>>(src, dst, n);
  };
  auto gemm = [&](const __bf16* A, const __bf16* W, const float* bias,
                  const float* addend, float* C, __bf16* Cbf,
                  int Mm, int Nn, int Kk, int act) {
    k_wmma_gemm<<<dim3(Nn / 128, Mm / 64), blk, 0, stream>>>(
        A, W, bias, addend, C, Cbf, Mm, Nn, Kk, act);
  };

  // 0) weights -> bf16 (each call; deterministic)
  cvt(gate_w,      gw_b,  D * D);
  cvt(state_in_w,  siw_b, KD * D);
  cvt(state_out_w, sow_b, D * KD);
  cvt(mlp_up_w,    upw_b, HID * D);
  cvt(mlp_down_w,  dnw_b, D * HID);

  // 1) xn = rms_norm(x)
  k_rmsnorm<<<dim3(M), blk, 0, stream>>>(x, xn_f, xn_b);
  // 2) gate_lin = xn @ gate_w^T  (bias applied inside k_u)
  gemm(xn_b, gw_b, nullptr, nullptr, gate_l, nullptr, M, D, D, 0);
  // 3) u = silu(conv4(xn)) * sigmoid(gate_lin + gate_b)
  k_u<<<dim3((M * D) / 256), blk, 0, stream>>>(xn_f, gate_l, gate_b, conv_w,
                                               u_f, u_b);
  // 4) inp = u @ state_in_w^T
  gemm(u_b, siw_b, nullptr, nullptr, inp_f, nullptr, M, KD, D, 0);
  // 5) SSM decay scan
  k_ssm<<<dim3(16384 / 256), blk, 0, stream>>>(inp_f, decay_logit, s_b);
  // 6) long = s_seq @ state_out_w^T
  gemm(s_b, sow_b, nullptr, nullptr, long_f, nullptr, M, D, KD, 0);
  // 7) skinny heads from u
  k_heads<<<dim3(M), blk, 0, stream>>>(u_f, read_bits_w, write_bits_w, qkey_w,
                                       wkey_w, value_w, write_gate_w, write_gate_b,
                                       gate_long_w, gate_long_b, gate_mem_w, gate_mem_b,
                                       rb_f, wb_f, qk_f, wk_f, wv_f, wg_f, gl_f, gm_f);
  // 8) hash memory: init + sequential scan
  k_meminit<<<dim3(262144 / 256), blk, 0, stream>>>(memkv, memlast);
  k_memscan<<<dim3(1), blk, 0, stream>>>(rb_f, wb_f, qk_f, wk_f, wv_f, wg_f,
                                         mem_k, mem_v, memlast, reads);
  // 9) y = x + gl*long + gm*(reads @ mem_out_w^T)
  k_ycomb<<<dim3((M * D) / 256), blk, 0, stream>>>(x, long_f, reads, mem_out_w,
                                                   gl_f, gm_f, y_f);
  // 10) yn = rms_norm(y) (bf16 only)
  k_rmsnorm<<<dim3(M), blk, 0, stream>>>(y_f, nullptr, yn_b);
  // 11) t1 = silu(yn @ mlp_up_w^T + mlp_up_b) -> bf16
  gemm(yn_b, upw_b, mlp_up_b, nullptr, nullptr, t1_b, M, HID, D, 1);
  // 12) out = y + t1 @ mlp_down_w^T + mlp_down_b
  gemm(t1_b, dnw_b, mlp_down_b, y_f, (float*)d_out, nullptr, M, D, HID, 0);
}
